// PlyFP4Linear_10986526343176
// MI455X (gfx1250) — compile-verified, matrix operand order checked by reading, not run on hardware
//
#include <hip/hip_runtime.h>
#include <stdint.h>

typedef __attribute__((ext_vector_type(16))) _Float16 v16h;
typedef __attribute__((ext_vector_type(8)))  _Float16 v8h;
typedef __attribute__((ext_vector_type(8)))  float    v8f;
typedef __attribute__((ext_vector_type(4)))  int      v4i;

#define MDIM 4096
#define NDIM 4096
#define KDIM 4096
#define BM 128
#define BN 128
#define BK 32
#define LDP 40   // padded LDS row stride in halves (80B -> conflict-free row access)
#define NT (KDIM / BK)

static __device__ __forceinline__ v16h combine16(v8h lo, v8h hi) {
    v16h r;
#pragma unroll
    for (int i = 0; i < 8; ++i) { r[i] = lo[i]; r[8 + i] = hi[i]; }
    return r;
}

// Stage one K-slice: A via async global->LDS DMA (ASYNCcnt), B via LUT dequant.
static __device__ __forceinline__ void stage_tiles(
    const _Float16* __restrict__ x, const int* __restrict__ pw,
    const unsigned int* lutp, _Float16* At, _Float16* Bt,
    int rowBase, int colBase, int k0, int tid)
{
    // ---- A: 128x32 fp16, 512 16B chunks, 2 async DMA per thread ----
#pragma unroll
    for (int i = 0; i < 2; ++i) {
        int c  = tid + i * 256;
        int r  = c >> 2;
        int kc = (c & 3) * 8;
        uint64_t gaddr = (uint64_t)(uintptr_t)(x + (size_t)(rowBase + r) * KDIM + k0 + kc);
        unsigned laddr = (unsigned)(uintptr_t)(At + r * LDP + kc);
        asm volatile("global_load_async_to_lds_b128 %0, %1, off"
                     :: "v"(laddr), "v"(gaddr) : "memory");
    }

    // ---- B: dequant 128x32 fp16 from packed fp4; half-row per thread ----
    int n    = tid >> 1;
    int half = tid & 1;
    const int* src = pw + (size_t)(colBase + n) * (KDIM / 2) + (k0 >> 1) + half * 8;
    v4i p0 = *(const v4i*)(src);
    v4i p1 = *(const v4i*)(src + 4);
    unsigned int d[8];
#pragma unroll
    for (int j = 0; j < 4; ++j) d[j]     = lutp[p0[j] & 255];
#pragma unroll
    for (int j = 0; j < 4; ++j) d[4 + j] = lutp[p1[j] & 255];
    v4i o0 = { (int)d[0], (int)d[1], (int)d[2], (int)d[3] };
    v4i o1 = { (int)d[4], (int)d[5], (int)d[6], (int)d[7] };
    *(v4i*)(&Bt[n * LDP + half * 16])     = o0;
    *(v4i*)(&Bt[n * LDP + half * 16 + 8]) = o1;
}

__global__ __launch_bounds__(256)
void fp4lut_gemm_wmma(const _Float16* __restrict__ x,
                      const int*      __restrict__ pw,
                      const _Float16* __restrict__ lut,
                      _Float16*       __restrict__ y) {
    __shared__ __align__(16) _Float16 Atile[2][BM * LDP];
    __shared__ __align__(16) _Float16 Btile[2][BN * LDP];
    __shared__ unsigned int lutpair[256];   // byte -> packed {odd_k_f16, even_k_f16}

    const int tid  = threadIdx.x;
    const int lane = tid & 31;
    const int wave = tid >> 5;
    const int wm   = wave >> 2;   // 0..1  -> 64-row strip
    const int wn   = wave & 3;    // 0..3  -> 32-col strip

    const int rowBase = blockIdx.y * BM;
    const int colBase = blockIdx.x * BN;

    // One-time LUT pair table: high nibble = even k (low half of dword).
    {
        const unsigned short* lu = (const unsigned short*)lut;
        unsigned int hi = lu[tid >> 4];
        unsigned int lo = lu[tid & 15];
        lutpair[tid] = (lo << 16) | hi;
    }
    __syncthreads();              // lut table visible before first dequant

    v8f acc[4][2] = {};

    const int lm   = lane & 15;
    const int kodd = (lane >> 4) * 8;     // 0 for lanes 0-15, 8 for lanes 16-31

    int cur = 0;
    stage_tiles(x, pw, lutpair, Atile[0], Btile[0], rowBase, colBase, 0, tid);

    for (int kt = 0; kt < NT; ++kt) {
        // our async DMA stores to LDS must land before the group barrier
        asm volatile("s_wait_asynccnt 0x0" ::: "memory");
        __syncthreads();          // buf[cur] fully staged by all waves

        if (kt + 1 < NT)          // overlap next stage with this compute
            stage_tiles(x, pw, lutpair, Atile[cur ^ 1], Btile[cur ^ 1],
                        rowBase, colBase, (kt + 1) * BK, tid);

        // ---- fragments: lane = row(A)/col(B), K pattern {0..7,16..23}+kodd ----
        const _Float16* At = Atile[cur];
        const _Float16* Bt = Btile[cur];
        v16h afrag[4];
#pragma unroll
        for (int mt = 0; mt < 4; ++mt) {
            int r = wm * 64 + mt * 16 + lm;
            v8h lo = *(const v8h*)(&At[r * LDP + kodd]);
            v8h hi = *(const v8h*)(&At[r * LDP + kodd + 16]);
            afrag[mt] = combine16(lo, hi);
        }
        v16h bfrag[2];
#pragma unroll
        for (int nt = 0; nt < 2; ++nt) {
            int c = wn * 32 + nt * 16 + lm;
            v8h lo = *(const v8h*)(&Bt[c * LDP + kodd]);
            v8h hi = *(const v8h*)(&Bt[c * LDP + kodd + 16]);
            bfrag[nt] = combine16(lo, hi);
        }

#pragma unroll
        for (int mt = 0; mt < 4; ++mt)
#pragma unroll
            for (int nt = 0; nt < 2; ++nt)
                acc[mt][nt] = __builtin_amdgcn_wmma_f32_16x16x32_f16(
                    false, afrag[mt], false, bfrag[nt],
                    (short)0, acc[mt][nt], false, false);

        cur ^= 1;
    }

    // ---- epilogue: C/D layout -> VGPR i holds M = i + 8*(lane>=16), N = lane&15 ----
    const int rOff = (lane >> 4) * 8;
#pragma unroll
    for (int mt = 0; mt < 4; ++mt) {
#pragma unroll
        for (int nt = 0; nt < 2; ++nt) {
            int col = colBase + wn * 32 + nt * 16 + lm;
            int rb  = rowBase + wm * 64 + mt * 16 + rOff;
#pragma unroll
            for (int i = 0; i < 8; ++i) {
                y[(size_t)(rb + i) * NDIM + col] = (_Float16)acc[mt][nt][i];
            }
        }
    }
}

extern "C" void kernel_launch(void* const* d_in, const int* in_sizes, int n_in,
                              void* d_out, int out_size, void* d_ws, size_t ws_size,
                              hipStream_t stream) {
    const _Float16* x   = (const _Float16*)d_in[0];
    const int*      pw  = (const int*)d_in[1];
    const _Float16* lut = (const _Float16*)d_in[2];
    _Float16*       y   = (_Float16*)d_out;

    dim3 grid(NDIM / BN, MDIM / BM);
    dim3 block(256);
    fp4lut_gemm_wmma<<<grid, block, 0, stream>>>(x, pw, lut, y);
}
